// Attention_old_90211493085279
// MI455X (gfx1250) — matrix operand order repaired and checked by reading, not compiled, not run on hardware
//
#include <hip/hip_runtime.h>

#define B_ 2
#define N_ 2048
#define C_ 1024
#define H_ 16
#define HD_ 64
#define WBAND 8

typedef __attribute__((ext_vector_type(16))) __bf16 v16bf;
typedef __attribute__((ext_vector_type(8))) float v8f;

union Frag { uint4 u[2]; v16bf v; };

__device__ __forceinline__ unsigned short f2bf(float f) {
  unsigned u = __float_as_uint(f);
  u += 0x7FFFu + ((u >> 16) & 1u);
  return (unsigned short)(u >> 16);
}
__device__ __forceinline__ float bflo(unsigned v) { return __uint_as_float(v << 16); }
__device__ __forceinline__ float bfhi(unsigned v) { return __uint_as_float(v & 0xFFFF0000u); }

// -------- CDNA5 async global->LDS copy (ASYNCcnt-tracked), with sync fallback
#if defined(__gfx1250__) && __has_builtin(__builtin_amdgcn_global_load_async_to_lds_b128)
#define ASYNC_LDS 1
#else
#define ASYNC_LDS 0
#endif

#if ASYNC_LDS
typedef int v4i_ __attribute__((vector_size(4 * sizeof(int))));
typedef __attribute__((address_space(1))) v4i_* gp16_t;
typedef __attribute__((address_space(3))) v4i_* lp16_t;
#endif

__device__ __forceinline__ void cp16(const unsigned short* g, unsigned short* l) {
#if ASYNC_LDS
  __builtin_amdgcn_global_load_async_to_lds_b128((gp16_t)(g), (lp16_t)(l), 0, 0);
#else
  *(uint4*)l = *(const uint4*)g;
#endif
}

__device__ __forceinline__ void wait_async() {
#if ASYNC_LDS
#if __has_builtin(__builtin_amdgcn_s_wait_asynccnt)
  __builtin_amdgcn_s_wait_asynccnt(0);
#else
  asm volatile("s_wait_asynccnt 0x0" ::: "memory");
#endif
#endif
}

// ---------------------------------------------------------------- convert (x)
__global__ __launch_bounds__(256) void cvt_f32_bf16(const float* __restrict__ in,
                                                    unsigned short* __restrict__ out, int n) {
  int i = blockIdx.x * 256 + threadIdx.x;
  if (i < n) out[i] = f2bf(in[i]);
}

// ---------------------------------------------------------------- convert + transpose (weights)
// in: [K][Nc] fp32  ->  out: [Nc][K] bf16   (LDS-tiled 32x32)
__global__ __launch_bounds__(256) void cvtT_f32_bf16(const float* __restrict__ in,
                                                     unsigned short* __restrict__ out,
                                                     int K, int Nc) {
  __shared__ unsigned short tile[32][33];
  const int n0 = blockIdx.x * 32;
  const int k0 = blockIdx.y * 32;
  const int tx = threadIdx.x & 31;
  const int ty = threadIdx.x >> 5;  // 0..7
  #pragma unroll
  for (int r = ty; r < 32; r += 8)
    tile[r][tx] = f2bf(in[(size_t)(k0 + r) * Nc + n0 + tx]);
  __syncthreads();
  #pragma unroll
  for (int r = ty; r < 32; r += 8)
    out[(size_t)(n0 + r) * K + k0 + tx] = tile[tx][r];
}

// ---------------------------------------------------------------- GEMM (bf16 WMMA)
// C[M x Ncols] = A[M x K] * B[K x Ncols] with B given pre-transposed Bt[Ncols x K].
// Double-buffered LDS tiles filled by async global->LDS copies.
// mode 0: scatter bf16 result into q/k/v [B,H,N,HD] layouts
// mode 1: add bias, write fp32 to out
#define APITCH 40
#define BPITCH 40

__global__ __launch_bounds__(256) void gemm_wmma_bf16(
    const unsigned short* __restrict__ A,
    const unsigned short* __restrict__ Bt,
    int K, int Ncols, int mode,
    unsigned short* __restrict__ qb, unsigned short* __restrict__ kb,
    unsigned short* __restrict__ vb,
    const float* __restrict__ bias, float* __restrict__ out)
{
  __shared__ unsigned short As[2][128 * APITCH];  // [row][k] bf16
  __shared__ unsigned short Bs[2][128 * BPITCH];  // [col][k] bf16

  const int t = threadIdx.x;
  const int lane = t & 31;
  const int wid = t >> 5;
  const int waveM = (wid & 1) * 64;
  const int waveN = (wid >> 1) * 32;
  const int lcol = lane & 15;
  const int lhi = (lane >> 4) & 1;

  const int blockCol = blockIdx.x * 128;
  const int blockRow = blockIdx.y * 128;

  v8f acc[4][2];
  #pragma unroll
  for (int i = 0; i < 4; i++)
    #pragma unroll
    for (int j = 0; j < 2; j++)
      #pragma unroll
      for (int g = 0; g < 8; g++) acc[i][j][g] = 0.0f;

  // staging: 2 threads per 32-elem row for both A and Bt (identical pattern)
  const int ar = t >> 1;
  const int ah = (t & 1) * 16;
  const unsigned short* gA  = A  + (size_t)(blockRow + ar) * K + ah;
  const unsigned short* gBt = Bt + (size_t)(blockCol + ar) * K + ah;

  const int nk = K >> 5;

  // prologue: fill buffer 0
  cp16(gA,      &As[0][ar * APITCH + ah]);
  cp16(gA + 8,  &As[0][ar * APITCH + ah + 8]);
  cp16(gBt,     &Bs[0][ar * BPITCH + ah]);
  cp16(gBt + 8, &Bs[0][ar * BPITCH + ah + 8]);

  for (int ks = 0; ks < nk; ks++) {
    const int cur = ks & 1;
    wait_async();
    __syncthreads();
    if (ks + 1 < nk) {
      const int k0 = (ks + 1) << 5;
      cp16(gA + k0,      &As[1 - cur][ar * APITCH + ah]);
      cp16(gA + k0 + 8,  &As[1 - cur][ar * APITCH + ah + 8]);
      cp16(gBt + k0,     &Bs[1 - cur][ar * BPITCH + ah]);
      cp16(gBt + k0 + 8, &Bs[1 - cur][ar * BPITCH + ah + 8]);
      if (ks + 2 < nk) {
        __builtin_prefetch(gA + ((ks + 2) << 5), 0, 1);   // global_prefetch_b8
        __builtin_prefetch(gBt + ((ks + 2) << 5), 0, 1);
      }
    }

    // ---- build fragments per ISA 7.12.2 wave32 layouts
    Frag af[4], bfr[2];
    #pragma unroll
    for (int mt = 0; mt < 4; mt++) {
      int r = waveM + mt * 16 + lcol;
      int e0 = lhi ? 8 : 0;   // lanes<16: K0..7/K16..23 ; lanes>=16: K8..15/K24..31
      af[mt].u[0] = *(const uint4*)&As[cur][r * APITCH + e0];
      af[mt].u[1] = *(const uint4*)&As[cur][r * APITCH + e0 + 16];
    }
    #pragma unroll
    for (int nt = 0; nt < 2; nt++) {
      int n = waveN + nt * 16 + lcol;
      int e0 = lhi ? 8 : 0;   // lanes<16: K0..7/K16..23 ; lanes>=16: K8..15/K24..31
      bfr[nt].u[0] = *(const uint4*)&Bs[cur][n * BPITCH + e0];
      bfr[nt].u[1] = *(const uint4*)&Bs[cur][n * BPITCH + e0 + 16];
    }
    #pragma unroll
    for (int mt = 0; mt < 4; mt++)
      #pragma unroll
      for (int nt = 0; nt < 2; nt++)
        acc[mt][nt] = __builtin_amdgcn_wmma_f32_16x16x32_bf16(
            false, af[mt].v, false, bfr[nt].v, (short)0, acc[mt][nt], false, false);
    __syncthreads();
  }

  // ---- epilogue (C/D layout: VGPR g = row g (lanes<16) / g+8 (lanes>=16), col = lane%16)
  if (mode == 0) {
    #pragma unroll
    for (int mt = 0; mt < 4; mt++) {
      #pragma unroll
      for (int nt = 0; nt < 2; nt++) {
        int col3 = blockCol + waveN + nt * 16 + lcol;
        int part = col3 >> 10;
        int c = col3 & 1023;
        int h = c >> 6, d = c & 63;
        unsigned short* dst = (part == 0) ? qb : (part == 1) ? kb : vb;
        #pragma unroll
        for (int g = 0; g < 8; g++) {
          int grow = blockRow + waveM + mt * 16 + g + lhi * 8;
          int b = grow >> 11, nr = grow & 2047;
          dst[(((size_t)b * H_ + h) * N_ + nr) * HD_ + d] = f2bf(acc[mt][nt][g]);
        }
      }
    }
  } else {
    #pragma unroll
    for (int mt = 0; mt < 4; mt++) {
      #pragma unroll
      for (int nt = 0; nt < 2; nt++) {
        int gcol = blockCol + waveN + nt * 16 + lcol;
        float bv = bias[gcol];
        #pragma unroll
        for (int g = 0; g < 8; g++) {
          int grow = blockRow + waveM + mt * 16 + g + lhi * 8;
          out[(size_t)grow * Ncols + gcol] = acc[mt][nt][g] + bv;
        }
      }
    }
  }
}

// ---------------------------------------------------------------- banded attention
// One block: one (b,h), 64 query rows. 4 lanes per query (16 dims each).
__global__ __launch_bounds__(256) void band_attn(
    const unsigned short* __restrict__ qb,
    const unsigned short* __restrict__ kb,
    const unsigned short* __restrict__ vb,
    unsigned short* __restrict__ attb)
{
  const int KP = 72;
  __shared__ unsigned short Ks[80 * 72];
  __shared__ unsigned short Vs[80 * 72];
  const int bh = blockIdx.x;
  const int q0 = blockIdx.y * 64;
  const int t = threadIdx.x;
  const int kLo = q0 - WBAND;
  const size_t base = (size_t)bh * N_ * HD_;

  for (int idx = t; idx < 80 * 8; idx += 256) {
    int jj = idx >> 3;
    int cc = (idx & 7) * 8;
    int g = kLo + jj;
    uint4 kq = make_uint4(0, 0, 0, 0), vq = make_uint4(0, 0, 0, 0);
    if (g >= 0 && g < N_) {
      kq = *(const uint4*)&kb[base + (size_t)g * HD_ + cc];
      vq = *(const uint4*)&vb[base + (size_t)g * HD_ + cc];
    }
    *(uint4*)&Ks[jj * KP + cc] = kq;
    *(uint4*)&Vs[jj * KP + cc] = vq;
  }
  __syncthreads();

  const int qi = q0 + (t >> 2);
  const int d0 = (t & 3) * 16;
  float qf[16];
  {
    union { uint4 u; unsigned v[4]; } lo, hi;
    lo.u = *(const uint4*)&qb[base + (size_t)qi * HD_ + d0];
    hi.u = *(const uint4*)&qb[base + (size_t)qi * HD_ + d0 + 8];
    #pragma unroll
    for (int i = 0; i < 4; i++) {
      qf[2 * i] = bflo(lo.v[i]);     qf[2 * i + 1] = bfhi(lo.v[i]);
      qf[8 + 2 * i] = bflo(hi.v[i]); qf[8 + 2 * i + 1] = bfhi(hi.v[i]);
    }
  }
  const float scale = 0.125f;  // HD^-0.5
  float s[17];
  const int jj0 = qi - q0;
  #pragma unroll
  for (int o = 0; o < 17; o++) {
    int j = qi - WBAND + o;
    const unsigned short* kr = &Ks[(jj0 + o) * KP + d0];
    union { uint4 u; unsigned v[4]; } lo, hi;
    lo.u = *(const uint4*)&kr[0];
    hi.u = *(const uint4*)&kr[8];
    float p = 0.f;
    #pragma unroll
    for (int i = 0; i < 4; i++) {
      p += qf[2 * i] * bflo(lo.v[i]);         p += qf[2 * i + 1] * bfhi(lo.v[i]);
      p += qf[8 + 2 * i] * bflo(hi.v[i]);     p += qf[8 + 2 * i + 1] * bfhi(hi.v[i]);
    }
    p += __shfl_xor(p, 1, 32);
    p += __shfl_xor(p, 2, 32);
    s[o] = (j >= 0 && j < N_) ? p * scale : -1e30f;
  }
  float m = s[0];
  #pragma unroll
  for (int o = 1; o < 17; o++) m = fmaxf(m, s[o]);
  float sum = 0.f;
  #pragma unroll
  for (int o = 0; o < 17; o++) { s[o] = __expf(s[o] - m); sum += s[o]; }
  float inv = 1.0f / sum;
  float acc[16];
  #pragma unroll
  for (int i = 0; i < 16; i++) acc[i] = 0.f;
  #pragma unroll
  for (int o = 0; o < 17; o++) {
    float p = s[o] * inv;
    const unsigned short* vr = &Vs[(jj0 + o) * KP + d0];
    union { uint4 u; unsigned v[4]; } lo, hi;
    lo.u = *(const uint4*)&vr[0];
    hi.u = *(const uint4*)&vr[8];
    #pragma unroll
    for (int i = 0; i < 4; i++) {
      acc[2 * i] += p * bflo(lo.v[i]);         acc[2 * i + 1] += p * bfhi(lo.v[i]);
      acc[8 + 2 * i] += p * bflo(hi.v[i]);     acc[8 + 2 * i + 1] += p * bfhi(hi.v[i]);
    }
  }
  const int b = bh >> 4, h = bh & 15;
  union { uint4 u; unsigned v[4]; } o0, o1;
  #pragma unroll
  for (int i = 0; i < 4; i++) {
    o0.v[i] = (unsigned)f2bf(acc[2 * i]) | ((unsigned)f2bf(acc[2 * i + 1]) << 16);
    o1.v[i] = (unsigned)f2bf(acc[8 + 2 * i]) | ((unsigned)f2bf(acc[8 + 2 * i + 1]) << 16);
  }
  size_t orow = ((size_t)b * N_ + qi) * C_ + h * HD_ + d0;
  *(uint4*)&attb[orow] = o0.u;
  *(uint4*)&attb[orow + 8] = o1.u;
}

// ---------------------------------------------------------------- launch
extern "C" void kernel_launch(void* const* d_in, const int* in_sizes, int n_in,
                              void* d_out, int out_size, void* d_ws, size_t ws_size,
                              hipStream_t stream) {
  const float* x     = (const float*)d_in[0];
  const float* Wqkv  = (const float*)d_in[1];
  const float* Wproj = (const float*)d_in[2];
  const float* bproj = (const float*)d_in[3];
  float* out = (float*)d_out;

  char* w = (char*)d_ws;
  const size_t MBY = 1ull << 20;
  unsigned short* xb     = (unsigned short*)(w + 0 * MBY);   // 8 MB  [B*N][C] bf16
  unsigned short* wqkvT  = (unsigned short*)(w + 8 * MBY);   // 6 MB  [3C][C] bf16 (transposed)
  unsigned short* wprojT = (unsigned short*)(w + 14 * MBY);  // 2 MB  [C][C] bf16 (transposed)
  unsigned short* qb     = (unsigned short*)(w + 16 * MBY);  // 8 MB  [B,H,N,HD] bf16
  unsigned short* kb     = (unsigned short*)(w + 24 * MBY);  // 8 MB
  unsigned short* vb     = (unsigned short*)(w + 32 * MBY);  // 8 MB
  unsigned short* attb   = (unsigned short*)(w + 40 * MBY);  // 8 MB  [B*N][C] bf16

  const int nx = B_ * N_ * C_;
  cvt_f32_bf16<<<dim3((nx + 255) / 256), dim3(256), 0, stream>>>(x, xb, nx);
  cvtT_f32_bf16<<<dim3((3 * C_) / 32, C_ / 32), dim3(256), 0, stream>>>(Wqkv, wqkvT, C_, 3 * C_);
  cvtT_f32_bf16<<<dim3(C_ / 32, C_ / 32), dim3(256), 0, stream>>>(Wproj, wprojT, C_, C_);

  // QKV projection: [4096 x 1024] x [1024 x 3072] -> q/k/v bf16 [B,H,N,HD]
  gemm_wmma_bf16<<<dim3((3 * C_) / 128, (B_ * N_) / 128), dim3(256), 0, stream>>>(
      xb, wqkvT, C_, 3 * C_, 0, qb, kb, vb, nullptr, nullptr);

  // banded softmax attention -> attb bf16 [B*N, C]
  band_attn<<<dim3(B_ * H_, N_ / 64), dim3(256), 0, stream>>>(qb, kb, vb, attb);

  // output projection: [4096 x 1024] x [1024 x 1024] + bias -> fp32 d_out
  gemm_wmma_bf16<<<dim3(C_ / 128, (B_ * N_) / 128), dim3(256), 0, stream>>>(
      attb, wprojT, C_, C_, 1, nullptr, nullptr, nullptr, bproj, out);
}
